// MultiHeadAttention_16630113370910
// MI455X (gfx1250) — matrix operand order repaired and checked by reading, not compiled, not run on hardware
//
#include <hip/hip_runtime.h>

// ---------------------------------------------------------------------------
// CDNA5 (gfx1250, wave32) WMMA bf16 multi-head attention forward.
// Pipeline: f32->bf16 convert -> QKV GEMMs (async-LDS double buffered)
//           -> flash attention (async-LDS K/Q tiles) -> out GEMM+bias
// ---------------------------------------------------------------------------

typedef __attribute__((ext_vector_type(16))) __bf16 v16bf;
typedef __attribute__((ext_vector_type(8))) float v8f;

union FragU {
  v16bf v;
  unsigned int u[8];
};

__device__ __forceinline__ unsigned short f2bf(float f) {
  unsigned int u = __float_as_uint(f);
  u += 0x7fffu + ((u >> 16) & 1u);  // round-to-nearest-even
  return (unsigned short)(u >> 16);
}

// Async global->LDS copy of 16 bytes per lane (CDNA5 ASYNCcnt path, no VGPR
// data round-trip). lds_dst flat shared ptr: addr[31:0] is the LDS offset.
__device__ __forceinline__ void async_copy16(void* lds_dst, const void* gsrc) {
  unsigned int ldsa = (unsigned int)(unsigned long long)lds_dst;
  asm volatile("global_load_async_to_lds_b128 %0, %1, off"
               :
               : "v"(ldsa), "v"(gsrc)
               : "memory");
}

__device__ __forceinline__ void wait_async0() {
  asm volatile("s_wait_asynccnt 0x0" ::: "memory");
}

// A-matrix 16x32 bf16 fragment from LDS tile (row-major, stride ldk elems).
// Lanes 0-15 row M=lane, dwords 0-7 = K 0..15;
// lanes 16-31 same rows, K 16..31.
__device__ __forceinline__ v16bf load_frag_a(const unsigned short* smem, int row,
                                             int kbase, int ldk, int lane) {
  const int r = row + (lane & 15);
  const int hi = (lane >> 4) & 1;
  const int ka = kbase + (hi ? 16 : 0);
  const int kb = ka + 8;
  FragU f;
  const unsigned int* pa = (const unsigned int*)(smem + r * ldk + ka);
  const unsigned int* pb = (const unsigned int*)(smem + r * ldk + kb);
#pragma unroll
  for (int i = 0; i < 4; ++i) {
    f.u[i] = pa[i];
    f.u[4 + i] = pb[i];
  }
  return f.v;
}

// B-matrix 32x16 bf16 fragment. smem holds B TRANSPOSED: Bt[n][k], stride ldk.
// Lane holds col n = lane&15; lanes 0-15 K 0..15, lanes 16-31 K 16..31.
__device__ __forceinline__ v16bf load_frag_b(const unsigned short* smem, int col,
                                             int kbase, int ldk, int lane) {
  const int c = col + (lane & 15);
  const int k0 = kbase + ((lane & 16) ? 16 : 0);
  FragU f;
  const unsigned int* p = (const unsigned int*)(smem + c * ldk + k0);
#pragma unroll
  for (int i = 0; i < 8; ++i) f.u[i] = p[i];
  return f.v;
}

__device__ __forceinline__ v8f wmma_bf16(v16bf a, v16bf b, v8f c) {
  // (neg_a, A, neg_b, B, c_mod, C, reuse_a, reuse_b)
  return __builtin_amdgcn_wmma_f32_16x16x32_bf16(false, a, false, b, (short)0, c,
                                                 false, false);
}

// ---------------------------------------------------------------------------
// fp32 -> bf16 conversion
// ---------------------------------------------------------------------------
__global__ void f32_to_bf16_kernel(const float* __restrict__ in,
                                   unsigned short* __restrict__ out, int n) {
  int i = blockIdx.x * blockDim.x + threadIdx.x;
  if (i < n) out[i] = f2bf(in[i]);
}

// ---------------------------------------------------------------------------
// C = A(MxK) * B(KxN) [+ bias], bf16 inputs, f32 accum.
// Block: 256 threads (8 waves). Tile 128x128, K-step 32, double-buffered LDS.
// A tile staged with global_load_async_to_lds_b128; B tile staged through
// VGPRs (needs bf16-pair transpose split). Next tile fills while WMMAs run.
// ---------------------------------------------------------------------------
template <bool OUT_F32>
__global__ __launch_bounds__(256) void gemm_bf16_wmma(
    const unsigned short* __restrict__ A, const unsigned short* __restrict__ B,
    void* __restrict__ C, const float* __restrict__ bias, int M, int N, int K) {
  __shared__ alignas(16) unsigned short sA[2][128 * 32];  // [row][k]
  __shared__ alignas(16) unsigned short sB[2][128 * 32];  // transposed [n][k]
  const int tid = threadIdx.x;
  const int lane = tid & 31;
  const int wave = tid >> 5;
  const int wm = wave >> 2;  // 0..1
  const int wn = wave & 3;   // 0..3
  const long bm = (long)blockIdx.y * 128;
  const long bn = (long)blockIdx.x * 128;

  auto stageA = [&](int k0, int buf) {
    // 128x32 bf16 = 512 x 16B chunks, 2 per thread, async to LDS
#pragma unroll
    for (int i = 0; i < 2; ++i) {
      int c = tid + i * 256;
      int r = c >> 2, q = (c & 3) * 8;
      async_copy16(&sA[buf][r * 32 + q], A + (size_t)(bm + r) * K + k0 + q);
    }
  };
  auto stageB = [&](int k0, int buf) {
#pragma unroll
    for (int i = 0; i < 8; ++i) {
      int d = tid * 8 + i;
      int kk = d >> 6, c = d & 63;  // [32][64] dwords of global B
      unsigned int dv =
          *((const unsigned int*)(B + (size_t)(k0 + kk) * N + bn) + c);
      int n = c * 2;
      sB[buf][n * 32 + kk] = (unsigned short)(dv & 0xffffu);
      sB[buf][(n + 1) * 32 + kk] = (unsigned short)(dv >> 16);
    }
  };

  v8f acc[4][2];
#pragma unroll
  for (int i = 0; i < 4; ++i)
#pragma unroll
    for (int j = 0; j < 2; ++j)
#pragma unroll
      for (int e = 0; e < 8; ++e) acc[i][j][e] = 0.0f;

  stageA(0, 0);
  stageB(0, 0);
  wait_async0();
  __syncthreads();

  int buf = 0;
  for (int k0 = 0; k0 < K; k0 += 32) {
    // Kick off next tile into the other buffer while we compute.
    if (k0 + 32 < K) {
      stageA(k0 + 32, buf ^ 1);
      stageB(k0 + 32, buf ^ 1);
    }

    v16bf af[4], bfr[2];
#pragma unroll
    for (int i = 0; i < 4; ++i)
      af[i] = load_frag_a(sA[buf], wm * 64 + i * 16, 0, 32, lane);
#pragma unroll
    for (int j = 0; j < 2; ++j)
      bfr[j] = load_frag_b(sB[buf], wn * 32 + j * 16, 0, 32, lane);
#pragma unroll
    for (int i = 0; i < 4; ++i)
#pragma unroll
      for (int j = 0; j < 2; ++j) acc[i][j] = wmma_bf16(af[i], bfr[j], acc[i][j]);

    wait_async0();    // next-tile async copies landed in LDS
    __syncthreads();  // everyone done reading current buffer
    buf ^= 1;
  }

  // Epilogue. C layout: col = lane&15, row = vgpr + 8*(lane>=16).
  const int cl = lane & 15;
  const int rh = (lane & 16) ? 8 : 0;
#pragma unroll
  for (int j = 0; j < 2; ++j) {
    const long c = bn + wn * 32 + j * 16 + cl;
    const float bv = bias ? bias[c] : 0.0f;
#pragma unroll
    for (int i = 0; i < 4; ++i) {
#pragma unroll
      for (int e = 0; e < 8; ++e) {
        const long r = bm + wm * 64 + i * 16 + rh + e;
        const float val = acc[i][j][e] + bv;
        if (OUT_F32)
          ((float*)C)[(size_t)r * N + c] = val;
        else
          ((unsigned short*)C)[(size_t)r * N + c] = f2bf(val);
      }
    }
  }
}

// ---------------------------------------------------------------------------
// Flash attention, causal, head_dim=64.
// Block: 128 threads (4 waves) handles one (batch, head, 64-query tile).
// Q/K tiles are async-copied to LDS; V goes through VGPRs (transpose).
// ---------------------------------------------------------------------------
__global__ __launch_bounds__(128) void attn_fwd(
    const unsigned short* __restrict__ Qb, const unsigned short* __restrict__ Kb,
    const unsigned short* __restrict__ Vb, unsigned short* __restrict__ Ctx,
    int S) {
  __shared__ alignas(16) unsigned short sQ[64 * 64];     // [q][d]
  __shared__ alignas(16) unsigned short sK[64 * 64];     // [key][d] == Bt(QK^T)
  __shared__ alignas(16) unsigned short sV[64 * 64];     // [d][key] == Bt(PV)
  __shared__ alignas(16) unsigned short sP[4][16 * 64];  // per-wave P [q][key]

  const int tid = threadIdx.x, lane = tid & 31, wave = tid >> 5;
  const int qt = blockIdx.x, h = blockIdx.y, b = blockIdx.z;
  const int q0 = qt * 64;
  const size_t bOff = (size_t)b * S * 768;
  const unsigned short* Qg = Qb + bOff + (size_t)q0 * 768 + h * 64;

  // Async-load Q tile (64x64 bf16 = 512 x 16B chunks, 4 per thread)
#pragma unroll
  for (int i = 0; i < 4; ++i) {
    int c = tid + i * 128;
    int r = c >> 3, q = (c & 7) * 8;
    async_copy16(&sQ[r * 64 + q], Qg + (size_t)r * 768 + q);
  }
  wait_async0();
  __syncthreads();
  const v16bf aQ0 = load_frag_a(sQ, wave * 16, 0, 64, lane);
  const v16bf aQ1 = load_frag_a(sQ, wave * 16, 32, 64, lane);

  float mrow[8], lrow[8];
  v8f accO[4];
#pragma unroll
  for (int t = 0; t < 4; ++t)
#pragma unroll
    for (int e = 0; e < 8; ++e) accO[t][e] = 0.0f;
#pragma unroll
  for (int e = 0; e < 8; ++e) {
    mrow[e] = -__builtin_inff();
    lrow[e] = 0.0f;
  }

  const int cl = lane & 15;
  const int rh = (lane & 16) ? 8 : 0;
  const int qrow0 = q0 + wave * 16 + rh;

  for (int jt = 0; jt <= qt; ++jt) {
    const int k0g = jt * 64;
    const unsigned short* Kg = Kb + bOff + (size_t)k0g * 768 + h * 64;
    const unsigned short* Vg = Vb + bOff + (size_t)k0g * 768 + h * 64;
    // K tile: async copy (row-major, used directly as Bt for Q*K^T)
#pragma unroll
    for (int i = 0; i < 4; ++i) {
      int c = tid + i * 128;
      int r = c >> 3, q = (c & 7) * 8;
      async_copy16(&sK[r * 64 + q], Kg + (size_t)r * 768 + q);
    }
    // V tile: VGPR transpose into [d][key]
#pragma unroll
    for (int i = 0; i < 16; ++i) {
      int d = tid * 16 + i;
      int r = d >> 5, c = d & 31;
      unsigned int dv = *((const unsigned int*)(Vg + (size_t)r * 768) + c);
      sV[(2 * c) * 64 + r] = (unsigned short)(dv & 0xffffu);
      sV[(2 * c + 1) * 64 + r] = (unsigned short)(dv >> 16);
    }
    wait_async0();
    __syncthreads();

    // S = Q * K^T : 16x64 per wave, 4 N-tiles x 2 K-steps
    v8f s[4];
#pragma unroll
    for (int t = 0; t < 4; ++t) {
#pragma unroll
      for (int e = 0; e < 8; ++e) s[t][e] = 0.0f;
      s[t] = wmma_bf16(aQ0, load_frag_b(sK, t * 16, 0, 64, lane), s[t]);
      s[t] = wmma_bf16(aQ1, load_frag_b(sK, t * 16, 32, 64, lane), s[t]);
    }

    // Scale by 1/sqrt(64), causal mask on the diagonal tile
    const bool diag = (jt == qt);
#pragma unroll
    for (int t = 0; t < 4; ++t)
#pragma unroll
      for (int e = 0; e < 8; ++e) {
        float v = s[t][e] * 0.125f;
        if (diag && (k0g + t * 16 + cl > qrow0 + e)) v = -__builtin_inff();
        s[t][e] = v;
      }

    // Online softmax: row max / rescale / row sum (16-lane reductions)
    float fexp[8];
#pragma unroll
    for (int e = 0; e < 8; ++e) {
      float mx = s[0][e];
#pragma unroll
      for (int t = 1; t < 4; ++t) mx = fmaxf(mx, s[t][e]);
#pragma unroll
      for (int off = 8; off >= 1; off >>= 1)
        mx = fmaxf(mx, __shfl_xor(mx, off, 16));
      const float mn = fmaxf(mrow[e], mx);
      fexp[e] = (mrow[e] == -__builtin_inff()) ? 0.0f : __expf(mrow[e] - mn);
      mrow[e] = mn;
    }
#pragma unroll
    for (int e = 0; e < 8; ++e) {
      float rs = 0.0f;
#pragma unroll
      for (int t = 0; t < 4; ++t) {
        const float p = __expf(s[t][e] - mrow[e]);
        s[t][e] = p;
        rs += p;
      }
#pragma unroll
      for (int off = 8; off >= 1; off >>= 1) rs += __shfl_xor(rs, off, 16);
      lrow[e] = lrow[e] * fexp[e] + rs;
#pragma unroll
      for (int t = 0; t < 4; ++t) accO[t][e] *= fexp[e];
    }

    // Stage P (C layout -> LDS -> A layout), wave-private region
    unsigned short* pw = sP[wave];
#pragma unroll
    for (int t = 0; t < 4; ++t)
#pragma unroll
      for (int e = 0; e < 8; ++e)
        pw[(rh + e) * 64 + t * 16 + cl] = f2bf(s[t][e]);

    const v16bf aP0 = load_frag_a(pw, 0, 0, 64, lane);
    const v16bf aP1 = load_frag_a(pw, 0, 32, 64, lane);
#pragma unroll
    for (int t = 0; t < 4; ++t) {
      accO[t] = wmma_bf16(aP0, load_frag_b(sV, t * 16, 0, 64, lane), accO[t]);
      accO[t] = wmma_bf16(aP1, load_frag_b(sV, t * 16, 32, 64, lane), accO[t]);
    }
    __syncthreads();  // before next tile overwrites sK/sV
  }

  // Normalize and write ctx (bf16) back in (b, s, 768) layout
  unsigned short* Cg = Ctx + bOff + (size_t)q0 * 768 + h * 64;
#pragma unroll
  for (int e = 0; e < 8; ++e) {
    const float inv = 1.0f / lrow[e];
#pragma unroll
    for (int t = 0; t < 4; ++t)
      Cg[(size_t)(wave * 16 + rh + e) * 768 + t * 16 + cl] =
          f2bf(accO[t][e] * inv);
  }
}

// ---------------------------------------------------------------------------
// Launcher. Inputs: x, W_q, W_k, W_v, W_o, b_o (all fp32). Output: fp32.
// Workspace layout (bf16/ushort elems):
//   xb[8192*768] wq wk wv wo[768*768 ea] Q K V ctx[8192*768 ea]  (~68 MB)
// ---------------------------------------------------------------------------
extern "C" void kernel_launch(void* const* d_in, const int* in_sizes, int n_in,
                              void* d_out, int out_size, void* d_ws,
                              size_t ws_size, hipStream_t stream) {
  (void)in_sizes; (void)n_in; (void)out_size; (void)ws_size;
  const float* x = (const float*)d_in[0];
  const float* Wq = (const float*)d_in[1];
  const float* Wk = (const float*)d_in[2];
  const float* Wv = (const float*)d_in[3];
  const float* Wo = (const float*)d_in[4];
  const float* bo = (const float*)d_in[5];

  const int B = 2, S = 4096, D = 768;
  const int M = B * S;  // 8192
  const size_t MD = (size_t)M * D;
  const size_t DD = (size_t)D * D;

  unsigned short* ws = (unsigned short*)d_ws;
  unsigned short* xb = ws;
  unsigned short* wq = xb + MD;
  unsigned short* wk = wq + DD;
  unsigned short* wv = wk + DD;
  unsigned short* wo = wv + DD;
  unsigned short* Qb = wo + DD;
  unsigned short* Kb = Qb + MD;
  unsigned short* Vb = Kb + MD;
  unsigned short* Cx = Vb + MD;

  // Convert fp32 -> bf16
  {
    int n = (int)MD;
    f32_to_bf16_kernel<<<(n + 255) / 256, 256, 0, stream>>>(x, xb, n);
    int m = (int)DD;
    f32_to_bf16_kernel<<<(m + 255) / 256, 256, 0, stream>>>(Wq, wq, m);
    f32_to_bf16_kernel<<<(m + 255) / 256, 256, 0, stream>>>(Wk, wk, m);
    f32_to_bf16_kernel<<<(m + 255) / 256, 256, 0, stream>>>(Wv, wv, m);
    f32_to_bf16_kernel<<<(m + 255) / 256, 256, 0, stream>>>(Wo, wo, m);
  }

  // Q/K/V projections
  dim3 gg(D / 128, M / 128);  // (6, 64)
  gemm_bf16_wmma<false><<<gg, 256, 0, stream>>>(xb, wq, Qb, nullptr, M, D, D);
  gemm_bf16_wmma<false><<<gg, 256, 0, stream>>>(xb, wk, Kb, nullptr, M, D, D);
  gemm_bf16_wmma<false><<<gg, 256, 0, stream>>>(xb, wv, Vb, nullptr, M, D, D);

  // Causal flash attention
  attn_fwd<<<dim3(S / 64, 12, B), 128, 0, stream>>>(Qb, Kb, Vb, Cx, S);

  // Output projection with bias -> fp32
  gemm_bf16_wmma<true><<<gg, 256, 0, stream>>>(Cx, wo, (float*)d_out, bo, M, D, D);
}
